// Ernie4_5_VLTextAttention_81819126989327
// MI455X (gfx1250) — matrix operand order repaired and checked by reading, not compiled, not run on hardware
//
#include <hip/hip_runtime.h>
#include <hip/hip_bf16.h>
#include <math.h>

#define B_    2
#define S_    2048
#define HID_  2048
#define NQ_   16
#define NKV_  4
#define D_    128
#define QKV_N 3072
#define GROUPS (NQ_ / NKV_)

typedef __attribute__((ext_vector_type(16))) __bf16        v16bf;
typedef __attribute__((ext_vector_type(8)))  float         v8f;
typedef __attribute__((ext_vector_type(4)))  unsigned int  v4u;
typedef __attribute__((ext_vector_type(8)))  unsigned int  v8u;

struct PairU { v4u lo, hi; };

__device__ __forceinline__ v16bf make_frag(v4u lo, v4u hi) {
  PairU p{lo, hi};
  return __builtin_bit_cast(v16bf, p);
}

__device__ __forceinline__ unsigned short f2bf(float x) {
  unsigned int u = __builtin_bit_cast(unsigned int, x);
  u += 0x7FFFu + ((u >> 16) & 1u);           // round-to-nearest-even
  return (unsigned short)(u >> 16);
}

// A-matrix fragment: lane = M row (lane&15); half 0 holds K {k0..k0+7, k0+16..k0+23},
// half 1 holds K {k0+8..k0+15, k0+24..k0+31}. Source row-major [M][K], K-contiguous.
__device__ __forceinline__ v16bf load_frag_a(const unsigned short* base, int ld, int k0, int lane) {
  int m = lane & 15, h = lane >> 4;
  const unsigned short* p = base + (size_t)m * ld + k0 + h * 8;
  v4u lo = *reinterpret_cast<const v4u*>(p);
  v4u hi = *reinterpret_cast<const v4u*>(p + 16);
  return make_frag(lo, hi);
}

// B-matrix fragment: lane = N column (lane&15); half 0 holds K {k0..k0+7, k0+16..k0+23},
// half 1 holds K {k0+8..k0+15, k0+24..k0+31}. Source stored as [N][K] row-major (i.e. B^T), K-contiguous.
__device__ __forceinline__ v16bf load_frag_b(const unsigned short* base, int ld, int k0, int lane) {
  int n = lane & 15, h = lane >> 4;
  const unsigned short* p = base + (size_t)n * ld + k0 + h * 8;
  v4u lo = *reinterpret_cast<const v4u*>(p);
  v4u hi = *reinterpret_cast<const v4u*>(p + 16);
  return make_frag(lo, hi);
}

__device__ __forceinline__ v8f wmma_bf16(v16bf a, v16bf b, v8f c) {
  return __builtin_amdgcn_wmma_f32_16x16x32_bf16(false, a, false, b, (short)0, c, false, false);
}

// ---- Tensor Data Mover: async 2D tile (bf16) global -> LDS. Wave-uniform D# in SGPRs. ----
// tile_d0 = contiguous elements per row, tile_d1 = rows, stride_d0 = row pitch (elements).
__device__ __forceinline__ void tdm_load_2d(unsigned lds_off, const void* gptr,
                                            unsigned tile_d0, unsigned tile_d1,
                                            unsigned stride_d0) {
  unsigned long long ga = (unsigned long long)(uintptr_t)gptr;
  v4u g0;
  g0[0] = 1u;                                            // count=1, user mode
  g0[1] = lds_off;                                       // lds_addr (bytes)
  g0[2] = (unsigned)ga;                                  // global_addr[31:0]
  g0[3] = (unsigned)((ga >> 32) & 0x01FFFFFFu) | (2u << 30); // addr[56:32] | type=2
  v8u g1;
  g1[0] = 1u << 16;                                      // data_size=1 (2 bytes)
  g1[1] = (stride_d0 & 0xFFFFu) << 16;                   // tensor_dim0[15:0]
  g1[2] = (stride_d0 >> 16) | (0xFFFFu << 16);           // tensor_dim0[31:16] | tensor_dim1 lo
  g1[3] = (tile_d0 << 16);                               // tensor_dim1 hi=0 | tile_dim0
  g1[4] = tile_d1;                                       // tile_dim1 | tile_dim2=0
  g1[5] = stride_d0;                                     // tensor_dim0_stride[31:0]
  g1[6] = 0u;                                            // stride hi | dim1_stride lo
  g1[7] = 0u;
  asm volatile("tensor_load_to_lds %0, %1" :: "s"(g0), "s"(g1) : "memory");
}

// ---------------- elementwise fp32 -> bf16 ----------------
__global__ void f2bf_kernel(const float* __restrict__ src, unsigned short* __restrict__ dst, size_t n) {
  size_t i = (size_t)blockIdx.x * blockDim.x + threadIdx.x;
  size_t stride = (size_t)gridDim.x * blockDim.x;
  for (; i < n; i += stride) dst[i] = f2bf(src[i]);
}

// ---------------- GEMM: C[M,N] = A[M,K] * B[N,K]^T (bf16 in, fp32 out) ----------------
// 4 waves/block; each wave computes a 32x64 tile (2 A-frags x 4 B-frags, 8 accumulators).
__global__ void __launch_bounds__(128)
gemm_bf16(const unsigned short* __restrict__ A, const unsigned short* __restrict__ Bm,
          float* __restrict__ C, int M, int N, int K) {
  int lane = threadIdx.x & 31;
  int wave = threadIdx.x >> 5;
  int m0 = blockIdx.y * 128 + wave * 32;
  int n0 = blockIdx.x * 64;
  v8f acc0[4], acc1[4];
#pragma unroll
  for (int t = 0; t < 4; ++t) { acc0[t] = {}; acc1[t] = {}; }
  const unsigned short* A0 = A + (size_t)m0 * K;
  const unsigned short* A1 = A + (size_t)(m0 + 16) * K;
  const unsigned short* B0 = Bm + (size_t)(n0 +  0) * K;
  const unsigned short* B1 = Bm + (size_t)(n0 + 16) * K;
  const unsigned short* B2 = Bm + (size_t)(n0 + 32) * K;
  const unsigned short* B3 = Bm + (size_t)(n0 + 48) * K;
  for (int kk = 0; kk < K; kk += 32) {
    v16bf a0 = load_frag_a(A0, K, kk, lane);
    v16bf a1 = load_frag_a(A1, K, kk, lane);
    v16bf b0 = load_frag_b(B0, K, kk, lane);
    v16bf b1 = load_frag_b(B1, K, kk, lane);
    v16bf b2 = load_frag_b(B2, K, kk, lane);
    v16bf b3 = load_frag_b(B3, K, kk, lane);
    acc0[0] = wmma_bf16(a0, b0, acc0[0]);
    acc1[0] = wmma_bf16(a1, b0, acc1[0]);
    acc0[1] = wmma_bf16(a0, b1, acc0[1]);
    acc1[1] = wmma_bf16(a1, b1, acc1[1]);
    acc0[2] = wmma_bf16(a0, b2, acc0[2]);
    acc1[2] = wmma_bf16(a1, b2, acc1[2]);
    acc0[3] = wmma_bf16(a0, b3, acc0[3]);
    acc1[3] = wmma_bf16(a1, b3, acc1[3]);
  }
  int n = lane & 15, h = lane >> 4;
#pragma unroll
  for (int r = 0; r < 8; ++r) {
    size_t ma = (size_t)(m0 + r + 8 * h);
    size_t mb = ma + 16;
#pragma unroll
    for (int t = 0; t < 4; ++t) {
      C[ma * N + n0 + t * 16 + n] = acc0[t][r];
      C[mb * N + n0 + t * 16 + n] = acc1[t][r];
    }
  }
}

// ---------------- RoPE: qkv fp32 -> qr/kr bf16 (rotated) ----------------
__global__ void __launch_bounds__(64)
rope_kernel(const float* __restrict__ qkv, const long long* __restrict__ pos,
            unsigned short* __restrict__ qr, unsigned short* __restrict__ kr) {
  int j  = threadIdx.x;                 // pair index 0..63
  int blk = blockIdx.x;
  int hh = blk % (NQ_ + NKV_);
  int bs = blk / (NQ_ + NKV_);
  int s = bs % S_, b = bs / S_;
  const long long* p3 = pos + ((size_t)b * S_ + s) * 3;
  float pt = (float)p3[0], ph = (float)p3[1], pw = (float)p3[2];
  int fi; float pp;
  if (j < 44) { if (j & 1) { fi = 22 + (j >> 1); pp = pw; } else { fi = j >> 1; pp = ph; } }
  else        { fi = j; pp = pt; }
  float inv_f = __powf(500000.0f, -(float)(2 * fi) * (1.0f / 128.0f));
  float ang = pp * inv_f;
  float sn, c;
  __sincosf(ang, &sn, &c);
  size_t row = (size_t)b * S_ + s;
  if (hh < NQ_) {
    const float* src = qkv + row * QKV_N + hh * D_ + 2 * j;
    float x0 = src[0], x1 = src[1];
    unsigned short* dst = qr + (((size_t)b * NQ_ + hh) * S_ + s) * D_ + 2 * j;
    dst[0] = f2bf(x0 * c - x1 * sn);
    dst[1] = f2bf(x1 * c + x0 * sn);
  } else {
    int hk = hh - NQ_;
    const float* src = qkv + row * QKV_N + NQ_ * D_ + hk * D_ + 2 * j;
    float x0 = src[0], x1 = src[1];
    unsigned short* dst = kr + (((size_t)b * NKV_ + hk) * S_ + s) * D_ + 2 * j;
    dst[0] = f2bf(x0 * c - x1 * sn);
    dst[1] = f2bf(x1 * c + x0 * sn);
  }
}

// ---------------- V transpose: qkv fp32 [b,s,v(kvh,d)] -> vt bf16 [b,kvh,d,s] ----------------
__global__ void __launch_bounds__(512)
vtrans_kernel(const float* __restrict__ qkv, unsigned short* __restrict__ vt) {
  int t = threadIdx.x;                  // 0..511 = kvh*128 + d
  int bs = blockIdx.x;
  int s = bs % S_, b = bs / S_;
  int kvh = t >> 7, d = t & 127;
  float x = qkv[((size_t)b * S_ + s) * QKV_N + (NQ_ + NKV_) * D_ + t];
  vt[(((size_t)b * NKV_ + kvh) * D_ + d) * S_ + s] = f2bf(x);
}

// ---------------- Flash attention with TDM double-buffered K/V staging ----------------
// One wave per (b, head, 16-row q tile). Per 32-wide K chunk:
//   TDM prefetch next chunk's K (32x128) and V^T (128x32) tiles into LDS,
//   s_wait_tensorcnt on current chunk, QK^T (8 wmma), online softmax, P.V (8 wmma).
__global__ void __launch_bounds__(32)
attn_kernel(const unsigned short* __restrict__ qr, const unsigned short* __restrict__ kr,
            const unsigned short* __restrict__ vt, unsigned short* __restrict__ ao) {
  __shared__ unsigned short Klds[2][32 * 128];
  __shared__ unsigned short Vlds[2][128 * 32];
  __shared__ unsigned short Plds[16 * 32];
  int lane = threadIdx.x;
  int qt = blockIdx.x, h = blockIdx.y, b = blockIdx.z;
  int kvh = h / GROUPS;
  const unsigned short* qbase = qr + (((size_t)b * NQ_ + h)   * S_ + qt * 16) * D_;
  const unsigned short* kbase = kr + (((size_t)b * NKV_ + kvh) * S_) * D_;
  const unsigned short* vbase = vt + (((size_t)b * NKV_ + kvh) * D_) * S_;

  v16bf qf[4];
#pragma unroll
  for (int i = 0; i < 4; ++i) qf[i] = load_frag_a(qbase, D_, i * 32, lane);

  v8f o[8];
#pragma unroll
  for (int i = 0; i < 8; ++i) o[i] = {};
  float rmax[8], rsum[8];
#pragma unroll
  for (int r = 0; r < 8; ++r) { rmax[r] = -3.0e38f; rsum[r] = 0.0f; }

  int n = lane & 15, hf = lane >> 4;
  const float scaling = 0.08838834764831845f;     // D^-0.5
  int nchunks = (qt * 16 + 47) / 32;

  // prefetch chunk 0
  tdm_load_2d((unsigned)(uintptr_t)&Klds[0][0], kbase, /*d0*/128, /*d1*/32, /*stride*/D_);
  tdm_load_2d((unsigned)(uintptr_t)&Vlds[0][0], vbase, /*d0*/32, /*d1*/128, /*stride*/S_);

  for (int kc = 0; kc < nchunks; ++kc) {
    int ks = kc * 32;
    int buf = kc & 1;
    if (kc + 1 < nchunks) {
      int ks2 = ks + 32;
      tdm_load_2d((unsigned)(uintptr_t)&Klds[buf ^ 1][0], kbase + (size_t)ks2 * D_, 128, 32, D_);
      tdm_load_2d((unsigned)(uintptr_t)&Vlds[buf ^ 1][0], vbase + ks2, 32, 128, S_);
      asm volatile("s_wait_tensorcnt 0x2" ::: "memory");  // current chunk's 2 loads done
    } else {
      asm volatile("s_wait_tensorcnt 0x0" ::: "memory");
    }
    const unsigned short* Kc = &Klds[buf][0];
    const unsigned short* Vc = &Vlds[buf][0];

    v8f s0 = {}, s1 = {};
#pragma unroll
    for (int i = 0; i < 4; ++i) {
      v16bf kb0 = load_frag_b(Kc,            D_, i * 32, lane);
      v16bf kb1 = load_frag_b(Kc + 16 * D_,  D_, i * 32, lane);
      s0 = wmma_bf16(qf[i], kb0, s0);
      s1 = wmma_bf16(qf[i], kb1, s1);
    }
#pragma unroll
    for (int r = 0; r < 8; ++r) {
      int qg = qt * 16 + r + 8 * hf;
      float x0 = s0[r] * scaling; if (ks + n > qg)      x0 = -3.0e38f;
      float x1 = s1[r] * scaling; if (ks + 16 + n > qg) x1 = -3.0e38f;
      float cm = fmaxf(x0, x1);
#pragma unroll
      for (int off = 1; off < 16; off <<= 1) cm = fmaxf(cm, __shfl_xor(cm, off, 32));
      float mnew = fmaxf(rmax[r], cm);
      float sc = __expf(rmax[r] - mnew);
      rmax[r] = mnew;
      float p0 = __expf(x0 - mnew), p1 = __expf(x1 - mnew);
      float ps = p0 + p1;
#pragma unroll
      for (int off = 1; off < 16; off <<= 1) ps += __shfl_xor(ps, off, 32);
      rsum[r] = rsum[r] * sc + ps;
#pragma unroll
      for (int dt = 0; dt < 8; ++dt) o[dt][r] *= sc;
      int row = r + 8 * hf;
      Plds[row * 32 + n]      = f2bf(p0);
      Plds[row * 32 + 16 + n] = f2bf(p1);
    }
    __syncthreads();
    v16bf pf = load_frag_a(Plds, 32, 0, lane);
#pragma unroll
    for (int dt = 0; dt < 8; ++dt) {
      v16bf vf = load_frag_b(Vc + dt * 16 * 32, 32, 0, lane);
      o[dt] = wmma_bf16(pf, vf, o[dt]);
    }
    __syncthreads();
  }

  unsigned short* obase = ao + ((size_t)b * S_ + qt * 16) * (NQ_ * D_) + h * D_;
#pragma unroll
  for (int r = 0; r < 8; ++r) {
    float inv = 1.0f / rsum[r];
    size_t m = (size_t)(r + 8 * hf);
#pragma unroll
    for (int dt = 0; dt < 8; ++dt) {
      obase[m * (NQ_ * D_) + dt * 16 + n] = f2bf(o[dt][r] * inv);
    }
  }
}

extern "C" void kernel_launch(void* const* d_in, const int* in_sizes, int n_in,
                              void* d_out, int out_size, void* d_ws, size_t ws_size,
                              hipStream_t stream) {
  (void)in_sizes; (void)n_in; (void)out_size; (void)ws_size;
  const float*     hidden = (const float*)d_in[0];
  const long long* pos    = (const long long*)d_in[1];   // int64 position_ids
  const float*     wq     = (const float*)d_in[2];
  const float*     wk     = (const float*)d_in[3];
  const float*     wv     = (const float*)d_in[4];
  const float*     wo     = (const float*)d_in[5];

  char* ws = (char*)d_ws;
  size_t off = 0;
  auto alloc = [&](size_t bytes) -> char* {
    char* p = ws + off;
    off += (bytes + 255) & ~(size_t)255;
    return p;
  };
  unsigned short* hb   = (unsigned short*)alloc((size_t)B_ * S_ * HID_ * 2);       // 16 MB
  unsigned short* wqkv = (unsigned short*)alloc((size_t)QKV_N * HID_ * 2);         // 12 MB
  unsigned short* wob  = (unsigned short*)alloc((size_t)HID_ * NQ_ * D_ * 2);      //  8 MB
  float*          qkv  = (float*)         alloc((size_t)B_ * S_ * QKV_N * 4);      // 48 MB
  unsigned short* qr   = (unsigned short*)alloc((size_t)B_ * NQ_ * S_ * D_ * 2);   // 16 MB
  unsigned short* kr   = (unsigned short*)alloc((size_t)B_ * NKV_ * S_ * D_ * 2);  //  4 MB
  unsigned short* vt   = (unsigned short*)alloc((size_t)B_ * NKV_ * D_ * S_ * 2);  //  4 MB
  unsigned short* ao   = (unsigned short*)alloc((size_t)B_ * S_ * NQ_ * D_ * 2);   // 16 MB

  // fp32 -> bf16 conversions (weights packed as [wq; wk; wv] rows, K-contiguous)
  f2bf_kernel<<<1024, 256, 0, stream>>>(hidden, hb, (size_t)B_ * S_ * HID_);
  f2bf_kernel<<<1024, 256, 0, stream>>>(wq, wqkv, (size_t)NQ_ * D_ * HID_);
  f2bf_kernel<<<512, 256, 0, stream>>>(wk, wqkv + (size_t)NQ_ * D_ * HID_, (size_t)NKV_ * D_ * HID_);
  f2bf_kernel<<<512, 256, 0, stream>>>(wv, wqkv + (size_t)(NQ_ + NKV_) * D_ * HID_, (size_t)NKV_ * D_ * HID_);
  f2bf_kernel<<<1024, 256, 0, stream>>>(wo, wob, (size_t)HID_ * NQ_ * D_);

  // fused QKV projection: [4096,2048] x [3072,2048]^T -> [4096,3072] fp32
  dim3 g1(QKV_N / 64, (B_ * S_) / 128);
  gemm_bf16<<<g1, 128, 0, stream>>>(hb, wqkv, qkv, B_ * S_, QKV_N, HID_);

  // RoPE on Q,K -> bf16 layouts for attention; V -> transposed bf16
  rope_kernel<<<B_ * S_ * (NQ_ + NKV_), 64, 0, stream>>>(qkv, pos, qr, kr);
  vtrans_kernel<<<B_ * S_, NKV_ * D_, 0, stream>>>(qkv, vt);

  // causal GQA flash attention (TDM double-buffered)
  dim3 ga(S_ / 16, NQ_, B_);
  attn_kernel<<<ga, 32, 0, stream>>>(qr, kr, vt, ao);

  // output projection: [4096,2048] x [2048,2048]^T -> d_out fp32
  dim3 g2(HID_ / 64, (B_ * S_) / 128);
  gemm_bf16<<<g2, 128, 0, stream>>>(ao, wob, (float*)d_out, B_ * S_, HID_, NQ_ * D_);
}
